// Transducer_53085795778980
// MI455X (gfx1250) — matrix operand order repaired and checked by reading, not compiled, not run on hardware
//
#include <hip/hip_runtime.h>
#include <math.h>

#define B_  4
#define T_  192
#define U_  80
#define U1_ 81
#define DE_ 320
#define DD_ 320
#define H_  512
#define V_  1024

typedef __attribute__((ext_vector_type(16))) __bf16 v16bf;
typedef __attribute__((ext_vector_type(8)))  __bf16 v8bf;
typedef __attribute__((ext_vector_type(8)))  float  v8f;

// ---------------------------------------------------------------------------
// he[b,t,h] = enc[b,t,:] . W1[h, :DE]
// ---------------------------------------------------------------------------
__global__ void k_enc_proj(const float* __restrict__ enc,
                           const float* __restrict__ W1,
                           float* __restrict__ he) {
    __shared__ float row[DE_];
    const int r = blockIdx.x;                       // 0..B*T-1
    for (int i = threadIdx.x; i < DE_; i += blockDim.x) row[i] = enc[r * DE_ + i];
    __syncthreads();
    for (int h = threadIdx.x; h < H_; h += blockDim.x) {
        const float* w = W1 + h * (DE_ + DD_);
        float acc = 0.f;
#pragma unroll 4
        for (int k = 0; k < DE_; ++k) acc = fmaf(row[k], w[k], acc);
        he[r * H_ + h] = acc;
    }
}

// ---------------------------------------------------------------------------
// hdb1[b,u,h] = dec[b,u,:] . W1[h, DE:] + b1[h]
// ---------------------------------------------------------------------------
__global__ void k_dec_proj(const float* __restrict__ dec,
                           const float* __restrict__ W1,
                           const float* __restrict__ b1,
                           float* __restrict__ hdb1) {
    __shared__ float row[DD_];
    const int r = blockIdx.x;                       // 0..B*U1-1
    for (int i = threadIdx.x; i < DD_; i += blockDim.x) row[i] = dec[r * DD_ + i];
    __syncthreads();
    for (int h = threadIdx.x; h < H_; h += blockDim.x) {
        const float* w = W1 + h * (DE_ + DD_) + DE_;
        float acc = b1[h];
#pragma unroll 4
        for (int k = 0; k < DD_; ++k) acc = fmaf(row[k], w[k], acc);
        hdb1[r * H_ + h] = acc;
    }
}

// ---------------------------------------------------------------------------
// W2 f32 -> bf16 (row-major [V][H])
// ---------------------------------------------------------------------------
__global__ void k_cvt_w2(const float* __restrict__ W2, __bf16* __restrict__ w2bf) {
    const int i = blockIdx.x * blockDim.x + threadIdx.x;
    if (i < V_ * H_) w2bf[i] = (__bf16)W2[i];
}

// ---------------------------------------------------------------------------
// Fused joint network: logits GEMM (bf16 WMMA, f32 acc) + log-softmax gather.
// One block per (b,t) x 32-u tile. 8 waves; each wave: 32 x 128 x 512 with
// two M-subtiles sharing every B fragment (2x arithmetic intensity vs M=16).
// Emits lp_blank[b,t,u] and lp_y[b,t,u] only.
// ---------------------------------------------------------------------------
__global__ __launch_bounds__(256) void k_joint(
    const float*  __restrict__ he,
    const float*  __restrict__ hdb1,
    const __bf16* __restrict__ w2bf,
    const float*  __restrict__ b2,
    const int*    __restrict__ targets,
    float* __restrict__ lpb,
    float* __restrict__ lpy)
{
    constexpr int LDA = 520;                        // padded bf16 row stride (1040 B)
    __shared__ __align__(16) __bf16 As[32 * LDA];
    __shared__ float redM[32][8];
    __shared__ float redS[32][8];
    __shared__ float logZ[32];
    __shared__ float blankL[32];
    __shared__ float tgtL[32];
    __shared__ int   tcolS[32];

    const int u0   = blockIdx.x * 32;
    const int bt   = blockIdx.y;                    // b*T + t
    const int b    = bt / T_;
    const int tid  = threadIdx.x;
    const int lane = tid & 31;
    const int wave = tid >> 5;

    if (tid < 32) {
        const int u = u0 + tid;
        tcolS[tid] = (u < U_) ? targets[b * U_ + u] : -1;
    }

    // Stage A = tanh(he + hd + b1) as bf16 into LDS (32 rows x 512)
    const float* heRow = he + bt * H_;
    for (int idx = tid; idx < 32 * H_; idx += 256) {
        const int rr = idx >> 9;                    // H_ == 512
        const int hh = idx & (H_ - 1);
        int u = u0 + rr; if (u > U_) u = U_;        // clamp pad rows
        const float v = tanhf(heRow[hh] + hdb1[(b * U1_ + u) * H_ + hh]);
        As[rr * LDA + hh] = (__bf16)v;
    }
    __syncthreads();

    // WMMA GEMM: two 16-row M-subtiles x 128 cols (per wave) x 512 K
    const int nbase = wave * 128;
    const int ncol  = lane & 15;
    const int khalf = lane >> 4;

    v8f acc[2][8] = {};
    const __bf16* aBase0 = As + (lane & 15) * LDA + khalf * 16;
    const __bf16* aBase1 = aBase0 + 16 * LDA;

    for (int k0 = 0; k0 < H_; k0 += 32) {
        const v8bf a0lo = *(const v8bf*)(aBase0 + k0);
        const v8bf a0hi = *(const v8bf*)(aBase0 + k0 + 8);
        const v8bf a1lo = *(const v8bf*)(aBase1 + k0);
        const v8bf a1hi = *(const v8bf*)(aBase1 + k0 + 8);
        const v16bf a0 = __builtin_shufflevector(a0lo, a0hi,
                           0,1,2,3,4,5,6,7,8,9,10,11,12,13,14,15);
        const v16bf a1 = __builtin_shufflevector(a1lo, a1hi,
                           0,1,2,3,4,5,6,7,8,9,10,11,12,13,14,15);
        const __bf16* bPtr = w2bf + (nbase + ncol) * H_ + k0 + khalf * 16;
#pragma unroll
        for (int tile = 0; tile < 8; ++tile) {
            const v16bf bf = *(const v16bf*)(bPtr + tile * 16 * H_);
            acc[0][tile] = __builtin_amdgcn_wmma_f32_16x16x32_bf16(
                false, a0, false, bf, (short)0, acc[0][tile], false, false);
            acc[1][tile] = __builtin_amdgcn_wmma_f32_16x16x32_bf16(
                false, a1, false, bf, (short)0, acc[1][tile], false, false);
        }
    }

    // Epilogue: + b2, gather blank/target columns, rowwise logsumexp.
    float b2v[8];
#pragma unroll
    for (int tile = 0; tile < 8; ++tile) b2v[tile] = b2[nbase + tile * 16 + ncol];
#pragma unroll
    for (int m = 0; m < 2; ++m)
#pragma unroll
        for (int tile = 0; tile < 8; ++tile)
#pragma unroll
            for (int e = 0; e < 8; ++e) acc[m][tile][e] += b2v[tile];

    // register-resident gather (each (row,col) owned by exactly one lane)
#pragma unroll
    for (int m = 0; m < 2; ++m) {
#pragma unroll
        for (int tile = 0; tile < 8; ++tile) {
            const int col = nbase + tile * 16 + ncol;
#pragma unroll
            for (int e = 0; e < 8; ++e) {
                const int row = e + (khalf << 3) + (m << 4);
                if (col == 0)          blankL[row] = acc[m][tile][e];
                if (col == tcolS[row]) tgtL[row]   = acc[m][tile][e];
            }
        }
    }

    // per-wave partial max / sum-exp (rows e+0/8 in lane halves, +16 for m=1)
#pragma unroll
    for (int m = 0; m < 2; ++m) {
        float pm[8], ps[8];
#pragma unroll
        for (int e = 0; e < 8; ++e) {
            float mx = acc[m][0][e];
#pragma unroll
            for (int tile = 1; tile < 8; ++tile) mx = fmaxf(mx, acc[m][tile][e]);
            mx = fmaxf(mx, __shfl_xor(mx, 1, 32));
            mx = fmaxf(mx, __shfl_xor(mx, 2, 32));
            mx = fmaxf(mx, __shfl_xor(mx, 4, 32));
            mx = fmaxf(mx, __shfl_xor(mx, 8, 32));
            float s = 0.f;
#pragma unroll
            for (int tile = 0; tile < 8; ++tile) s += __expf(acc[m][tile][e] - mx);
            s += __shfl_xor(s, 1, 32);
            s += __shfl_xor(s, 2, 32);
            s += __shfl_xor(s, 4, 32);
            s += __shfl_xor(s, 8, 32);
            pm[e] = mx; ps[e] = s;
        }
        if ((lane & 15) == 0) {
#pragma unroll
            for (int e = 0; e < 8; ++e) {
                const int row = e + (khalf << 3) + (m << 4);
                redM[row][wave] = pm[e];
                redS[row][wave] = ps[e];
            }
        }
    }
    __syncthreads();

    if (tid < 32) {
        float M = redM[tid][0];
        for (int w = 1; w < 8; ++w) M = fmaxf(M, redM[tid][w]);
        float S = 0.f;
        for (int w = 0; w < 8; ++w) S += redS[tid][w] * __expf(redM[tid][w] - M);
        logZ[tid] = M + __logf(S);
    }
    __syncthreads();

    if (tid < 32) {
        const int u = u0 + tid;
        const float lz = logZ[tid];
        if (u < U1_) lpb[bt * U1_ + u] = blankL[tid] - lz;
        if (u < U_)  lpy[bt * U_  + u] = tgtL[tid]   - lz;
    }
}

// ---------------------------------------------------------------------------
// RNN-T alpha DP: anti-diagonal wavefront, one block (96 x 4 threads).
// alpha[t][u] = logaddexp(alpha[t-1][u] + lpb[t-1][u], alpha[t][u-1] + lpy[t][u-1])
// ---------------------------------------------------------------------------
__device__ inline float logaddexpf_dev(float a, float b) {
    const float m = fmaxf(a, b);
    if (m == -INFINITY) return -INFINITY;
    return m + log1pf(__expf(fminf(a, b) - m));
}

__global__ __launch_bounds__(384) void k_rnnt(
    const float* __restrict__ lpb,
    const float* __restrict__ lpy,
    const int*   __restrict__ ilen,
    const int*   __restrict__ tlen,
    float* __restrict__ out)
{
    __shared__ float val[B_][96];
    __shared__ float a0[B_][96];
    __shared__ float fin[B_];
    const int u = threadIdx.x;          // 0..95 (active: u <= 80)
    const int b = threadIdx.y;          // 0..3

    int ti = ilen[b] - 1; ti = ti < 0 ? 0 : (ti > T_ - 1 ? T_ - 1 : ti);
    int ui = tlen[b];     ui = ui < 0 ? 0 : (ui > U_     ? U_     : ui);

    if (u == 0) {                       // alpha0 = [0, cumsum(lp_y[b,0,:])]
        float acc = 0.f;
        a0[b][0] = 0.f;
        for (int j = 1; j <= U_; ++j) {
            acc += lpy[(b * T_ + 0) * U_ + (j - 1)];
            a0[b][j] = acc;
        }
        fin[b] = 0.f;
    }
    __syncthreads();

    float my = (u <= U_) ? a0[b][u] : -INFINITY;
    if (u <= U_) val[b][u] = my;
    if (u <= U_ && ti == 0 && u == ui) fin[b] = my + lpb[(b * T_ + 0) * U1_ + u];
    __syncthreads();

    for (int d = 1; d <= (T_ - 1) + U_; ++d) {
        const float left = (u >= 1 && u <= U_) ? val[b][u - 1] : -INFINITY;
        __syncthreads();
        const int t = d - u;
        if (u <= U_ && t >= 1 && t < T_) {
            const float base = my + lpb[(b * T_ + (t - 1)) * U1_ + u];
            const float diag = (u >= 1) ? (left + lpy[(b * T_ + t) * U_ + (u - 1)])
                                        : -INFINITY;
            my = logaddexpf_dev(base, diag);
            val[b][u] = my;
            if (t == ti && u == ui) fin[b] = my + lpb[(b * T_ + t) * U1_ + u];
        }
        __syncthreads();
    }

    if (u == 0 && b == 0) {
        float s = 0.f;
        for (int i = 0; i < B_; ++i) s += fin[i];
        out[0] = -s / (float)B_;
    }
}

// ---------------------------------------------------------------------------
extern "C" void kernel_launch(void* const* d_in, const int* in_sizes, int n_in,
                              void* d_out, int out_size, void* d_ws, size_t ws_size,
                              hipStream_t stream) {
    const float* enc     = (const float*)d_in[0];
    const float* dec     = (const float*)d_in[1];
    const float* W1      = (const float*)d_in[2];
    const float* b1      = (const float*)d_in[3];
    const float* W2      = (const float*)d_in[4];
    const float* b2      = (const float*)d_in[5];
    const int*   targets = (const int*)d_in[6];
    const int*   ilen    = (const int*)d_in[7];
    const int*   tlen    = (const int*)d_in[8];

    char* ws = (char*)d_ws;
    float*  he   = (float*)(ws + 0);          // 768*512*4   = 1,572,864
    float*  hdb1 = (float*)(ws + 1572864);    // 324*512*4   =   663,552
    __bf16* w2bf = (__bf16*)(ws + 2236416);   // 1024*512*2  = 1,048,576
    float*  lpb  = (float*)(ws + 3284992);    // 4*192*81*4  =   248,832
    float*  lpy  = (float*)(ws + 3533824);    // 4*192*80*4  =   245,760

    k_enc_proj<<<B_ * T_,  256, 0, stream>>>(enc, W1, he);
    k_dec_proj<<<B_ * U1_, 256, 0, stream>>>(dec, W1, b1, hdb1);
    k_cvt_w2<<<(V_ * H_) / 256, 256, 0, stream>>>(W2, w2bf);

    dim3 g((U1_ + 31) / 32, B_ * T_);         // 3 x 768 blocks
    k_joint<<<g, 256, 0, stream>>>(he, hdb1, w2bf, b2, targets, lpb, lpy);

    k_rnnt<<<1, dim3(96, B_), 0, stream>>>(lpb, lpy, ilen, tlen, (float*)d_out);
}
